// EdgeCycleSplitLayer_69372311765065
// MI455X (gfx1250) — matrix operand order repaired and checked by reading, not compiled, not run on hardware
//
#include <hip/hip_runtime.h>

// ---------------------------------------------------------------------------
// EdgeCycleSplitLayer for MI455X (gfx1250, wave32, WMMA).
// GEMMs run on the bf16 WMMA pipe with fp32 emulation via bf16 hi/lo split:
//   X*W ~= Xhi*Whi + Xhi*Wlo + Xlo*Whi   (3x v_wmma_f32_16x16x32_bf16)
// BatchNorm(train) is computed as per-block column partial sums reduced by a
// small deterministic kernel, then folded into y = relu(z*scale + shift).
// ---------------------------------------------------------------------------

typedef __attribute__((ext_vector_type(16))) __bf16 v16bf;
typedef __attribute__((ext_vector_type(8)))  __bf16 v8bf;
typedef __attribute__((ext_vector_type(8)))  float  v8f;
typedef __attribute__((ext_vector_type(4)))  float  v4f;

#define HDIM 128
#define BM   128
#define BK   32
#define LSTR 40   // LDS row stride in bf16 elems: 32 data + 8 pad (80 bytes)

__device__ __forceinline__ float relu_f(float x) { return x > 0.0f ? x : 0.0f; }

// MODE 0: A row = concat(edge_rep[eidx[r]], cycle_rep[cidx[r]])      (K=256)
// MODE 1: A row = alpha * A0[r] + A1[r]                              (K=128)
// MODE 2: A row = relu(A0[r] * inScale + inShift)                    (K=128)
template <int MODE>
__global__ __launch_bounds__(256)
void gemm_bn_kernel(const float* __restrict__ A0,
                    const float* __restrict__ A1,
                    const int*   __restrict__ eidx,
                    const int*   __restrict__ cidx,
                    const float* __restrict__ epsPtr,
                    const float* __restrict__ inScale,
                    const float* __restrict__ inShift,
                    const float* __restrict__ W,       // [K, 128] row-major
                    float*       __restrict__ Z,       // [M, 128] pre-BN out
                    float*       __restrict__ partials,// [nblocks, 256]
                    int M, int K)
{
    __shared__ __align__(16) __bf16 AsHi[BM * LSTR];
    __shared__ __align__(16) __bf16 AsLo[BM * LSTR];
    __shared__ __align__(16) __bf16 BsHi[HDIM * LSTR]; // N-major: [n][k]
    __shared__ __align__(16) __bf16 BsLo[HDIM * LSTR];
    __shared__ float colS[HDIM];
    __shared__ float colQ[HDIM];

    const int tid  = threadIdx.x;
    const int lane = tid & 31;
    const int wave = tid >> 5;
    const int half = (lane >> 4) & 1;
    const int lr   = lane & 15;
    const int rowBase = blockIdx.x * BM;

    float alpha = 1.0f;
    if (MODE == 1) alpha = 1.0f + epsPtr[0];

    v8f acc[8];
#pragma unroll
    for (int j = 0; j < 8; ++j) acc[j] = v8f{};

    const int ksteps = K / BK;
    for (int ks = 0; ks < ksteps; ++ks) {
        const int k0 = ks * BK;

        // ---- stage A tile [128 rows x 32 K] as bf16 hi/lo ----
#pragma unroll
        for (int i = 0; i < 4; ++i) {
            int c  = tid + i * 256;        // 0..1023 float4 chunks
            int r  = c >> 3;               // 0..127
            int kk = (c & 7) * 4;          // 0..28
            int gr = rowBase + r;
            v4f v = v4f{};
            if (gr < M) {
                if (MODE == 0) {
                    int kg = k0 + kk;
                    const float* src = (kg < HDIM)
                        ? (A0 + (size_t)eidx[gr] * HDIM + kg)
                        : (A1 + (size_t)cidx[gr] * HDIM + (kg - HDIM));
                    v = *(const v4f*)src;
                } else if (MODE == 1) {
                    v4f x = *(const v4f*)(A0 + (size_t)gr * HDIM + (k0 + kk));
                    v4f y = *(const v4f*)(A1 + (size_t)gr * HDIM + (k0 + kk));
                    v = alpha * x + y;
                } else {
                    v4f z = *(const v4f*)(A0 + (size_t)gr * HDIM + (k0 + kk));
#pragma unroll
                    for (int q = 0; q < 4; ++q)
                        v[q] = relu_f(z[q] * inScale[k0 + kk + q] + inShift[k0 + kk + q]);
                }
            }
#pragma unroll
            for (int q = 0; q < 4; ++q) {
                float  f  = v[q];
                __bf16 hb = (__bf16)f;
                __bf16 lb = (__bf16)(f - (float)hb);
                AsHi[r * LSTR + kk + q] = hb;
                AsLo[r * LSTR + kk + q] = lb;
            }
        }

        // ---- stage B tile: W[k0..k0+31][0..127] -> Bs[n][k] (transpose) ----
#pragma unroll
        for (int i = 0; i < 4; ++i) {
            int c  = tid + i * 256;        // 0..1023
            int kr = c >> 5;               // 0..31
            int n0 = (c & 31) * 4;         // 0..124
            v4f w = *(const v4f*)(W + (size_t)(k0 + kr) * HDIM + n0);
#pragma unroll
            for (int q = 0; q < 4; ++q) {
                float  f  = w[q];
                __bf16 hb = (__bf16)f;
                __bf16 lb = (__bf16)(f - (float)hb);
                BsHi[(n0 + q) * LSTR + kr] = hb;
                BsLo[(n0 + q) * LSTR + kr] = lb;
            }
        }
        __syncthreads();

        // ---- fragments + WMMA (wave owns 16-row strip, 8 col tiles) ----
        const int arow = wave * 16 + lr;
        v16bf aHi, aLo;
        {
            // A element q: K = q + 8*half (q<8), K = 16+(q-8)+8*half (q>=8)
            const __bf16* ph = &AsHi[arow * LSTR + half * 8];
            const __bf16* pl = &AsLo[arow * LSTR + half * 8];
            v8bf h0 = *(const v8bf*)(ph);
            v8bf h1 = *(const v8bf*)(ph + 16);
            v8bf l0 = *(const v8bf*)(pl);
            v8bf l1 = *(const v8bf*)(pl + 16);
#pragma unroll
            for (int q = 0; q < 8; ++q) {
                aHi[q] = h0[q]; aHi[8 + q] = h1[q];
                aLo[q] = l0[q]; aLo[8 + q] = l1[q];
            }
        }
#pragma unroll
        for (int j = 0; j < 8; ++j) {
            // B element q: K = q + 8*half (q<8), K = 16+(q-8)+8*half (q>=8)
            const __bf16* ph = &BsHi[(j * 16 + lr) * LSTR + half * 8];
            const __bf16* pl = &BsLo[(j * 16 + lr) * LSTR + half * 8];
            v16bf bHi, bLo;
            v8bf h0 = *(const v8bf*)(ph);
            v8bf h1 = *(const v8bf*)(ph + 16);
            v8bf l0 = *(const v8bf*)(pl);
            v8bf l1 = *(const v8bf*)(pl + 16);
#pragma unroll
            for (int q = 0; q < 8; ++q) {
                bHi[q] = h0[q]; bHi[8 + q] = h1[q];
                bLo[q] = l0[q]; bLo[8 + q] = l1[q];
            }
            acc[j] = __builtin_amdgcn_wmma_f32_16x16x32_bf16(false, aHi, false, bHi,
                                                             (short)0, acc[j], false, false);
            acc[j] = __builtin_amdgcn_wmma_f32_16x16x32_bf16(false, aHi, false, bLo,
                                                             (short)0, acc[j], false, false);
            acc[j] = __builtin_amdgcn_wmma_f32_16x16x32_bf16(false, aLo, false, bHi,
                                                             (short)0, acc[j], false, false);
        }
        __syncthreads();
    }

    // ---- epilogue: store Z, per-block column sums / sumsq ----
    if (tid < HDIM) { colS[tid] = 0.0f; colQ[tid] = 0.0f; }
    __syncthreads();

#pragma unroll
    for (int j = 0; j < 8; ++j) {
        int col = j * 16 + lr;
        float s = 0.0f, q = 0.0f;
#pragma unroll
        for (int r = 0; r < 8; ++r) {
            int grow = rowBase + wave * 16 + half * 8 + r; // C layout: M = 8*half + r
            float v = acc[j][r];
            s += v; q += v * v;
            if (grow < M) Z[(size_t)grow * HDIM + col] = v;
        }
        atomicAdd(&colS[col], s);   // ds_add_f32
        atomicAdd(&colQ[col], q);
    }
    __syncthreads();
    if (tid < HDIM) {
        partials[(size_t)blockIdx.x * 256 + tid]        = colS[tid];
        partials[(size_t)blockIdx.x * 256 + HDIM + tid] = colQ[tid];
    }
}

// Deterministic reduce of per-block partials -> folded BN scale/shift.
__global__ void bn_stats_kernel(const float* __restrict__ partials, int nblocks,
                                const float* __restrict__ g, const float* __restrict__ b,
                                float invM,
                                float* __restrict__ scale, float* __restrict__ shift)
{
    int t = threadIdx.x; // 128 threads
    float s = 0.0f, q = 0.0f;
    for (int bl = 0; bl < nblocks; ++bl) {
        s += partials[(size_t)bl * 256 + t];
        q += partials[(size_t)bl * 256 + HDIM + t];
    }
    float mean = s * invM;
    float var  = q * invM - mean * mean;
    float inv  = rsqrtf(var + 1e-5f);
    float sc   = g[t] * inv;
    scale[t] = sc;
    shift[t] = b[t] - mean * sc;
}

// lift_aggr[cidx[p]] += edge_rep[eidx[p]]
__global__ __launch_bounds__(256)
void scatter_x_kernel(const float* __restrict__ edge_rep,
                      const int* __restrict__ eidx, const int* __restrict__ cidx,
                      float* __restrict__ lift_aggr, int P)
{
    size_t tid = (size_t)blockIdx.x * 256 + threadIdx.x;
    size_t p = tid >> 7;
    int    h = (int)(tid & 127);
    if (p >= (size_t)P) return;
    float v = edge_rep[(size_t)eidx[p] * HDIM + h];
    atomicAdd(&lift_aggr[(size_t)cidx[p] * HDIM + h], v);
}

// aggr_cycle[cidx[p]] += relu(msgZ[p]*scale + shift)
__global__ __launch_bounds__(256)
void scatter_msg_kernel(const float* __restrict__ msgZ, const int* __restrict__ cidx,
                        const float* __restrict__ scale, const float* __restrict__ shift,
                        float* __restrict__ aggr_cycle, int P)
{
    size_t tid = (size_t)blockIdx.x * 256 + threadIdx.x;
    size_t p = tid >> 7;
    int    h = (int)(tid & 127);
    if (p >= (size_t)P) return;
    float m = relu_f(msgZ[p * HDIM + h] * scale[h] + shift[h]);
    atomicAdd(&aggr_cycle[(size_t)cidx[p] * HDIM + h], m);
}

// lvl_aggr[eidx[p]] += aggr_cycle[cidx[p]] - relu(msgZ[p]*scale + shift)
__global__ __launch_bounds__(256)
void scatter_back_kernel(const float* __restrict__ msgZ,
                         const int* __restrict__ eidx, const int* __restrict__ cidx,
                         const float* __restrict__ scale, const float* __restrict__ shift,
                         const float* __restrict__ aggr_cycle,
                         float* __restrict__ lvl_aggr, int P)
{
    size_t tid = (size_t)blockIdx.x * 256 + threadIdx.x;
    size_t p = tid >> 7;
    int    h = (int)(tid & 127);
    if (p >= (size_t)P) return;
    float m = relu_f(msgZ[p * HDIM + h] * scale[h] + shift[h]);
    float v = aggr_cycle[(size_t)cidx[p] * HDIM + h] - m;
    atomicAdd(&lvl_aggr[(size_t)eidx[p] * HDIM + h], v);
}

// buf[i] = relu(buf[i]*scale[i%128] + shift[i%128])
__global__ __launch_bounds__(256)
void bnrelu_inplace_kernel(float* __restrict__ buf,
                           const float* __restrict__ scale, const float* __restrict__ shift,
                           size_t n)
{
    size_t t = (size_t)blockIdx.x * 256 + threadIdx.x;
    if (t >= n) return;
    int h = (int)(t & 127);
    buf[t] = relu_f(buf[t] * scale[h] + shift[h]);
}

extern "C" void kernel_launch(void* const* d_in, const int* in_sizes, int n_in,
                              void* d_out, int out_size, void* d_ws, size_t ws_size,
                              hipStream_t stream)
{
    (void)in_sizes; (void)n_in; (void)out_size; (void)ws_size;
    const int E = 200000, C = 50000, P = 400000;

    const float* edge_rep  = (const float*)d_in[0];
    const float* cycle_rep = (const float*)d_in[1];
    const int*   eidx      = (const int*)d_in[2];
    const int*   cidx      = (const int*)d_in[3];
    const float* lvl1_W  = (const float*)d_in[4];
    const float* lvl1_g  = (const float*)d_in[5];
    const float* lvl1_b  = (const float*)d_in[6];
    const float* lvl2_W1 = (const float*)d_in[7];
    const float* lvl2_g1 = (const float*)d_in[8];
    const float* lvl2_b1 = (const float*)d_in[9];
    const float* lvl2_W2 = (const float*)d_in[10];
    const float* lvl2_g2 = (const float*)d_in[11];
    const float* lvl2_b2 = (const float*)d_in[12];
    const float* lift_W1 = (const float*)d_in[13];
    const float* lift_g1 = (const float*)d_in[14];
    const float* lift_b1 = (const float*)d_in[15];
    const float* lift_W2 = (const float*)d_in[16];
    const float* lift_g2 = (const float*)d_in[17];
    const float* lift_b2 = (const float*)d_in[18];
    const float* eps1 = (const float*)d_in[19];
    const float* eps2 = (const float*)d_in[20];

    const int blocksP = (P + BM - 1) / BM;  // 3125
    const int blocksE = (E + BM - 1) / BM;  // 1563
    const int blocksC = (C + BM - 1) / BM;  // 391

    // workspace layout (floats)
    float* ws       = (float*)d_ws;
    float* msgZ     = ws;                              // P*H (reused later)
    float* aggrCyc  = msgZ + (size_t)P * HDIM;         // C*H
    float* liftAggr = aggrCyc + (size_t)C * HDIM;      // C*H
    float* lvlAggr  = liftAggr + (size_t)C * HDIM;     // E*H
    float* partials = lvlAggr + (size_t)E * HDIM;      // blocksP*256
    float* ss       = partials + (size_t)blocksP * 256;
    float* scale1 = ss;        float* shift1 = ss + 128;
    float* scale2 = ss + 256;  float* shift2 = ss + 384;
    float* scale3 = ss + 512;  float* shift3 = ss + 640;
    float* scale4 = ss + 768;  float* shift4 = ss + 896;
    float* scale5 = ss + 1024; float* shift5 = ss + 1152;
    float* zE = msgZ;                        // reuse msgZ region after step 6
    float* zC = msgZ + (size_t)E * HDIM;

    float* edge_out  = (float*)d_out;
    float* cycle_out = edge_out + (size_t)E * HDIM;

    const int scatBlocks  = (P * HDIM) / 256;          // 200000
    const int bnrE = (E * HDIM) / 256;                 // 100000
    const int bnrC = (C * HDIM) / 256;                 // 25000

    // 1. zero aggregation buffers (contiguous region: aggrCyc, liftAggr, lvlAggr)
    hipMemsetAsync(aggrCyc, 0, (size_t)(2 * C + E) * HDIM * sizeof(float), stream);

    // 2. lift_aggr = segment_sum(edge_rep[eidx], cidx)
    scatter_x_kernel<<<scatBlocks, 256, 0, stream>>>(edge_rep, eidx, cidx, liftAggr, P);

    // 3. msgZ = concat(gather,gather) @ lvl1_W   (K=256) + stats
    gemm_bn_kernel<0><<<blocksP, 256, 0, stream>>>(edge_rep, cycle_rep, eidx, cidx,
        nullptr, nullptr, nullptr, lvl1_W, msgZ, partials, P, 2 * HDIM);
    bn_stats_kernel<<<1, 128, 0, stream>>>(partials, blocksP, lvl1_g, lvl1_b,
        1.0f / (float)P, scale1, shift1);

    // 5. lvl_aggr_cycle = segment_sum(relu(msgZ*s1+t1), cidx)
    scatter_msg_kernel<<<scatBlocks, 256, 0, stream>>>(msgZ, cidx, scale1, shift1, aggrCyc, P);
    // 6. lvl_aggr = segment_sum(aggrCyc[cidx] - msg, eidx)
    scatter_back_kernel<<<scatBlocks, 256, 0, stream>>>(msgZ, eidx, cidx, scale1, shift1,
        aggrCyc, lvlAggr, P);

    // 7-11. edge MLP: two GEMM+BN+ReLU blocks
    gemm_bn_kernel<1><<<blocksE, 256, 0, stream>>>(edge_rep, lvlAggr, nullptr, nullptr,
        eps1, nullptr, nullptr, lvl2_W1, zE, partials, E, HDIM);
    bn_stats_kernel<<<1, 128, 0, stream>>>(partials, blocksE, lvl2_g1, lvl2_b1,
        1.0f / (float)E, scale2, shift2);
    gemm_bn_kernel<2><<<blocksE, 256, 0, stream>>>(zE, nullptr, nullptr, nullptr,
        nullptr, scale2, shift2, lvl2_W2, edge_out, partials, E, HDIM);
    bn_stats_kernel<<<1, 128, 0, stream>>>(partials, blocksE, lvl2_g2, lvl2_b2,
        1.0f / (float)E, scale3, shift3);
    bnrelu_inplace_kernel<<<bnrE, 256, 0, stream>>>(edge_out, scale3, shift3, (size_t)E * HDIM);

    // 12-16. cycle MLP: two GEMM+BN+ReLU blocks
    gemm_bn_kernel<1><<<blocksC, 256, 0, stream>>>(cycle_rep, liftAggr, nullptr, nullptr,
        eps2, nullptr, nullptr, lift_W1, zC, partials, C, HDIM);
    bn_stats_kernel<<<1, 128, 0, stream>>>(partials, blocksC, lift_g1, lift_b1,
        1.0f / (float)C, scale4, shift4);
    gemm_bn_kernel<2><<<blocksC, 256, 0, stream>>>(zC, nullptr, nullptr, nullptr,
        nullptr, scale4, shift4, lift_W2, cycle_out, partials, C, HDIM);
    bn_stats_kernel<<<1, 128, 0, stream>>>(partials, blocksC, lift_g2, lift_b2,
        1.0f / (float)C, scale5, shift5);
    bnrelu_inplace_kernel<<<bnrC, 256, 0, stream>>>(cycle_out, scale5, shift5, (size_t)C * HDIM);
}
